// WindowAttention_15899968930103
// MI455X (gfx1250) — compile-verified, matrix operand order checked by reading, not run on hardware
//
#include <hip/hip_runtime.h>
#include <hip/hip_bf16.h>

typedef __attribute__((ext_vector_type(16))) _Float16 v16h;
typedef __attribute__((ext_vector_type(8)))  _Float16 v8h;
typedef __attribute__((ext_vector_type(8)))  float    v8f;

#define DEV __device__ __forceinline__

// ---- WMMA fragment helpers ------------------------------------------------
// CDNA5 16-bit A fragment (16x32, MxK), per ISA 7.12.2:
//   lane l holds row (l & 15); element j<8  -> K = (l>>4)*8 + j
//                              element j>=8 -> K = 16 + (l>>4)*8 + (j-8)
// B fragment uses the identical gather on B^T (row-major [n][k]).
DEV const _Float16* frag_ptr(const _Float16* base, int stride, int lane) {
    return base + (size_t)(lane & 15) * stride + (lane >> 4) * 8;
}

DEV v16h frag_from_f16(const _Float16* p) {
    v8h lo = *(const v8h*)(p);
    v8h hi = *(const v8h*)(p + 16);
    v16h r;
#pragma unroll
    for (int i = 0; i < 8; ++i) { r[i] = lo[i]; r[i + 8] = hi[i]; }
    return r;
}

DEV v8f wmma_f16(v16h a, v16h b, v8f c) {
    return __builtin_amdgcn_wmma_f32_16x16x32_f16(
        /*neg_a=*/false, a, /*neg_b=*/false, b,
        /*c_mod=*/(short)0, c, /*reuse_a=*/false, /*reuse_b=*/false);
}

// ---- problem constants ----------------------------------------------------
// B=2048, N=49 tokens, DIM=384, H=12, HD=32, NW=64, tokens padded to 64.
static constexpr int    kDim   = 384;
static constexpr int    kHeads = 12;
static constexpr int    kHd    = 32;
static constexpr int    kTokPad = 64;
static constexpr size_t kWEle  = (size_t)384 * 384;            // per weight matrix
static constexpr size_t kQkvEle = (size_t)2048 * 12 * 64 * 32; // padded per-head tiles
static constexpr int    kMTiles = 6272;                        // 2048*49/16

// ---- kernel 0: weights f32 -> f16 ----------------------------------------
__global__ void wa_prep_weights(const float* __restrict__ wq, const float* __restrict__ wk,
                                const float* __restrict__ wv, const float* __restrict__ wp,
                                _Float16* __restrict__ dst) {
    int idx = blockIdx.x * 256 + threadIdx.x;
    if (idx >= (int)(4 * kWEle)) return;
    int which = idx / (int)kWEle;
    int r     = idx - which * (int)kWEle;
    const float* s = (which == 0) ? wq : (which == 1) ? wk : (which == 2) ? wv : wp;
    dst[idx] = (_Float16)s[r];
}

// ---- kernel 1: fused QKV projection (x @ W^T + b), scatter to padded tiles
__global__ __launch_bounds__(256) void wa_qkv(
    const float* __restrict__ x,
    const _Float16* __restrict__ wqh, const _Float16* __restrict__ wkh,
    const _Float16* __restrict__ wvh,
    const float* __restrict__ bq, const float* __restrict__ bk,
    const float* __restrict__ bv,
    _Float16* __restrict__ qh, _Float16* __restrict__ kh,
    _Float16* __restrict__ vth)
{
    __shared__ __align__(16) _Float16 Asm[16 * 384];
    const int tid   = threadIdx.x;
    const int mtile = blockIdx.x;
    const int mode  = blockIdx.y;           // 0=q 1=k 2=v

    // stage 16 rows of x (f32 -> f16) into LDS
    const float* xrow = x + (size_t)mtile * 16 * kDim;
    for (int i = tid; i < 16 * kDim; i += 256) Asm[i] = (_Float16)xrow[i];
    __syncthreads();

    const _Float16* W    = (mode == 0) ? wqh : (mode == 1) ? wkh : wvh;
    const float*    bias = (mode == 0) ? bq  : (mode == 1) ? bk  : bv;

    const int lane = tid & 31;
    const int wave = tid >> 5;              // 8 waves, 3 n-tiles each => 384 cols
    const int n0   = wave * 48;

    v8f acc0 = {}, acc1 = {}, acc2 = {};
    for (int kb = 0; kb < kDim; kb += 32) {
        v16h a  = frag_from_f16(frag_ptr(Asm + kb, kDim, lane));
        v16h b0 = frag_from_f16(frag_ptr(W + (size_t)(n0 +  0) * kDim + kb, kDim, lane));
        v16h b1 = frag_from_f16(frag_ptr(W + (size_t)(n0 + 16) * kDim + kb, kDim, lane));
        v16h b2 = frag_from_f16(frag_ptr(W + (size_t)(n0 + 32) * kDim + kb, kDim, lane));
        acc0 = wmma_f16(a, b0, acc0);
        acc1 = wmma_f16(a, b1, acc1);
        acc2 = wmma_f16(a, b2, acc2);
    }

    const float scale = 0.17677669529663687f;  // 32^-0.5, folded into q
#pragma unroll
    for (int t = 0; t < 3; ++t) {
        v8f acc = (t == 0) ? acc0 : (t == 1) ? acc1 : acc2;
        const int   n  = n0 + t * 16 + (lane & 15);
        const float bn = bias[n];
        const int   hh = n >> 5, dd = n & 31;
#pragma unroll
        for (int r = 0; r < 8; ++r) {
            const int m   = mtile * 16 + (lane >> 4) * 8 + r;     // flat token, exact fit
            const int bi  = m / 49;
            const int tok = m - bi * 49;
            const float val = acc[r] + bn;
            if (mode == 0)
                qh[(((size_t)bi * kHeads + hh) * kTokPad + tok) * kHd + dd] =
                    (_Float16)(val * scale);
            else if (mode == 1)
                kh[(((size_t)bi * kHeads + hh) * kTokPad + tok) * kHd + dd] = (_Float16)val;
            else
                vth[(((size_t)bi * kHeads + hh) * kHd + dd) * kTokPad + tok] = (_Float16)val;
        }
    }
}

// ---- kernel 2: attention per (b, h) ---------------------------------------
__global__ __launch_bounds__(128) void wa_attn(
    const _Float16* __restrict__ qh, const _Float16* __restrict__ kh,
    const _Float16* __restrict__ vth,
    const float* __restrict__ table, const float* __restrict__ mask,
    _Float16* __restrict__ oh)
{
    __shared__ __align__(16) float    bias_sm[49 * 64];
    __shared__ __align__(16) _Float16 Psm[4 * 16 * 64];

    const int bh  = blockIdx.x;
    const int bi  = bh / kHeads;
    const int h   = bh - bi * kHeads;
    const int wdw = bi & 63;                // window index = b % 64
    const int tid = threadIdx.x;

    // stage rel-pos bias + mask; pad columns get -inf
    for (int idx = tid; idx < 49 * 64; idx += 128) {
        const int m = idx >> 6, n = idx & 63;
        float v;
        if (n < 49) {
            const int mi = m / 7, mj = m - mi * 7;
            const int ni = n / 7, nj = n - ni * 7;
            const int rel = (mi - ni + 6) * 13 + (mj - nj + 6);
            v = table[rel * kHeads + h] + mask[((size_t)wdw * 49 + m) * 49 + n];
        } else {
            v = -1e30f;
        }
        bias_sm[idx] = v;
    }
    __syncthreads();

    const int lane = tid & 31;
    const int wave = tid >> 5;              // wave == M-tile (rows wave*16 .. +15)
    const _Float16* qbh = qh  + (size_t)bh * kTokPad * kHd;
    const _Float16* kbh = kh  + (size_t)bh * kTokPad * kHd;
    const _Float16* vbh = vth + (size_t)bh * kHd * kTokPad;

    // S = (q*scale) @ k^T  : K = 32 exact, 4 n-tiles
    v16h aq = frag_from_f16(frag_ptr(qbh + (size_t)wave * 16 * kHd, kHd, lane));
    v8f s0 = {}, s1 = {}, s2 = {}, s3 = {};
    s0 = wmma_f16(aq, frag_from_f16(frag_ptr(kbh +  0 * kHd, kHd, lane)), s0);
    s1 = wmma_f16(aq, frag_from_f16(frag_ptr(kbh + 16 * kHd, kHd, lane)), s1);
    s2 = wmma_f16(aq, frag_from_f16(frag_ptr(kbh + 32 * kHd, kHd, lane)), s2);
    s3 = wmma_f16(aq, frag_from_f16(frag_ptr(kbh + 48 * kHd, kHd, lane)), s3);

    // bias + row softmax (each row lives in one 16-lane half of the wave)
    float p[4][8];
#pragma unroll
    for (int r = 0; r < 8; ++r) {
        const int m  = wave * 16 + (lane >> 4) * 8 + r;
        const int nb = lane & 15;
        float v0 = s0[r], v1 = s1[r], v2 = s2[r], v3 = s3[r];
        if (m < 49) {
            v0 += bias_sm[m * 64 +  0 + nb];
            v1 += bias_sm[m * 64 + 16 + nb];
            v2 += bias_sm[m * 64 + 32 + nb];
            v3 += bias_sm[m * 64 + 48 + nb];
        }
        float mx = fmaxf(fmaxf(v0, v1), fmaxf(v2, v3));
#pragma unroll
        for (int off = 1; off < 16; off <<= 1) mx = fmaxf(mx, __shfl_xor(mx, off, 16));
        v0 = __expf(v0 - mx); v1 = __expf(v1 - mx);
        v2 = __expf(v2 - mx); v3 = __expf(v3 - mx);
        float sm = v0 + v1 + v2 + v3;
#pragma unroll
        for (int off = 1; off < 16; off <<= 1) sm += __shfl_xor(sm, off, 16);
        const float inv = 1.0f / sm;
        p[0][r] = v0 * inv; p[1][r] = v1 * inv; p[2][r] = v2 * inv; p[3][r] = v3 * inv;
    }

    // C-layout -> A-layout via LDS (per-wave private 16x64 tile)
    _Float16* Pw = Psm + wave * 16 * 64;
#pragma unroll
    for (int t = 0; t < 4; ++t)
#pragma unroll
        for (int r = 0; r < 8; ++r)
            Pw[((lane >> 4) * 8 + r) * 64 + t * 16 + (lane & 15)] = (_Float16)p[t][r];
    __syncthreads();

    // out = P @ v : K = 64 (2 steps), N = 32 (2 tiles)
#pragma unroll
    for (int nt = 0; nt < 2; ++nt) {
        v8f o = {};
#pragma unroll
        for (int ks = 0; ks < 2; ++ks) {
            v16h a = frag_from_f16(frag_ptr(Pw + ks * 32, 64, lane));
            v16h b = frag_from_f16(
                frag_ptr(vbh + (size_t)(nt * 16) * kTokPad + ks * 32, kTokPad, lane));
            o = wmma_f16(a, b, o);
        }
#pragma unroll
        for (int r = 0; r < 8; ++r) {
            const int tok = wave * 16 + (lane >> 4) * 8 + r;
            if (tok < 49) {
                const int dd = nt * 16 + (lane & 15);
                oh[((size_t)bi * 49 + tok) * kDim + h * kHd + dd] = (_Float16)o[r];
            }
        }
    }
}

// ---- kernel 3: output projection (oh @ Wp^T + bp), fp32 out ---------------
__global__ __launch_bounds__(256) void wa_proj(
    const _Float16* __restrict__ oh, const _Float16* __restrict__ wph,
    const float* __restrict__ bp, float* __restrict__ out)
{
    __shared__ __align__(16) _Float16 Asm[16 * 384];
    const int tid   = threadIdx.x;
    const int mtile = blockIdx.x;

    const _Float16* arow = oh + (size_t)mtile * 16 * kDim;
    for (int i = tid * 8; i < 16 * kDim; i += 256 * 8)
        *(v8h*)(Asm + i) = *(const v8h*)(arow + i);
    __syncthreads();

    const int lane = tid & 31;
    const int wave = tid >> 5;
    const int n0   = wave * 48;

    v8f a0 = {}, a1 = {}, a2 = {};
    for (int kb = 0; kb < kDim; kb += 32) {
        v16h a = frag_from_f16(frag_ptr(Asm + kb, kDim, lane));
        a0 = wmma_f16(a, frag_from_f16(frag_ptr(wph + (size_t)(n0 +  0) * kDim + kb, kDim, lane)), a0);
        a1 = wmma_f16(a, frag_from_f16(frag_ptr(wph + (size_t)(n0 + 16) * kDim + kb, kDim, lane)), a1);
        a2 = wmma_f16(a, frag_from_f16(frag_ptr(wph + (size_t)(n0 + 32) * kDim + kb, kDim, lane)), a2);
    }

#pragma unroll
    for (int t = 0; t < 3; ++t) {
        v8f acc = (t == 0) ? a0 : (t == 1) ? a1 : a2;
        const int   n  = n0 + t * 16 + (lane & 15);
        const float bn = bp[n];
#pragma unroll
        for (int r = 0; r < 8; ++r) {
            const int m = mtile * 16 + (lane >> 4) * 8 + r;
            out[(size_t)m * kDim + n] = acc[r] + bn;
        }
    }
}

// ---- launcher -------------------------------------------------------------
extern "C" void kernel_launch(void* const* d_in, const int* in_sizes, int n_in,
                              void* d_out, int out_size, void* d_ws, size_t ws_size,
                              hipStream_t stream) {
    (void)in_sizes; (void)n_in; (void)out_size; (void)ws_size;
    const float* x     = (const float*)d_in[0];
    const float* mask  = (const float*)d_in[1];
    const float* Wq    = (const float*)d_in[2];
    const float* bq    = (const float*)d_in[3];
    const float* Wk    = (const float*)d_in[4];
    const float* bk    = (const float*)d_in[5];
    const float* Wv    = (const float*)d_in[6];
    const float* bv    = (const float*)d_in[7];
    const float* Wp    = (const float*)d_in[8];
    const float* bp    = (const float*)d_in[9];
    const float* table = (const float*)d_in[10];

    _Float16* ws  = (_Float16*)d_ws;
    _Float16* wqh = ws;
    _Float16* wkh = ws + 1 * kWEle;
    _Float16* wvh = ws + 2 * kWEle;
    _Float16* wph = ws + 3 * kWEle;
    _Float16* qh  = ws + 4 * kWEle;
    _Float16* kh  = qh + kQkvEle;
    _Float16* vth = kh + kQkvEle;
    _Float16* oh  = vth + kQkvEle;

    // zero padded q/k/v^T tiles (rows 49..63 stay zero / -inf'd in softmax)
    hipMemsetAsync(qh, 0, 3 * kQkvEle * sizeof(_Float16), stream);

    wa_prep_weights<<<(4 * (int)kWEle + 255) / 256, 256, 0, stream>>>(Wq, Wk, Wv, Wp, ws);
    wa_qkv<<<dim3(kMTiles, 3), 256, 0, stream>>>(x, wqh, wkh, wvh, bq, bk, bv, qh, kh, vth);
    wa_attn<<<2048 * kHeads, 128, 0, stream>>>(qh, kh, vth, table, mask, oh);
    wa_proj<<<kMTiles, 256, 0, stream>>>(oh, wph, bp, (float*)d_out);
}
